// Sae_20598663151877
// MI455X (gfx1250) — matrix operand order repaired and checked by reading, not compiled, not run on hardware
//
#include <hip/hip_runtime.h>
#include <stdint.h>

// ---------------------------------------------------------------------------
// SAE forward for MI455X (gfx1250, wave32, WMMA):
//   latents = x @ W_enc^T + b_enc        [N=16384, L=12288], D=768
//   top-20 per row -> sparse latents
//   recons  = sparse @ W_dec^T + b_dec   [N, D]
// Kernel 1: fused bf16-WMMA encoder GEMM + per-row top-K (latents never hit HBM)
//   - block = 64 rows of x (bf16 tile resident in LDS, 96 KB)
//   - 8 waves x 16 latents per 128-wide latent chunk; 4 M-tiles per wave
//   - B (W_enc) streamed from L2 via clause'd global_load_b128 (compiler
//     pipelines these under the WMMAs); A fragments fed by ds_load_b128
// Kernel 2: sparse decoder gather (W_dec 38 MB fully L2-resident in 192 MB L2)
// ---------------------------------------------------------------------------

#define SAE_N   16384
#define SAE_D   768
#define SAE_L   12288
#define SAE_K   20
#define MBLK    64          // rows of x per workgroup
#define NCHUNK  128         // latents per chunk (8 waves x 16)
#define NCHUNKS (SAE_L / NCHUNK)   // 96
#define D_DW    (SAE_D / 2)        // 384 packed-bf16 dwords per row

typedef __attribute__((ext_vector_type(16))) __bf16 v16bf;
typedef __attribute__((ext_vector_type(8)))  float  v8f;

union Frag16 { v16bf v; uint32_t u[8]; };

// round-to-nearest-even f32 -> bf16, packed pair
__device__ __forceinline__ uint32_t pk_bf16(float lo, float hi) {
    union { float f; uint32_t u; } a, b;
    a.f = lo; b.f = hi;
    uint32_t al = (a.u + 0x7FFFu + ((a.u >> 16) & 1u)) >> 16;
    uint32_t bh = (b.u + 0x7FFFu + ((b.u >> 16) & 1u)) >> 16;
    return al | (bh << 16);
}

// rare-path insert into per-row top-K list held in LDS
__device__ __forceinline__ void topk_insert(float v, int idx,
                                            float* tv, int* ti, float& rmin) {
    if (v > rmin) {
        int am = 0; float mv = tv[0];
        #pragma unroll
        for (int t = 1; t < SAE_K; ++t) {
            const float x = tv[t];
            if (x < mv) { mv = x; am = t; }
        }
        tv[am] = v; ti[am] = idx;
        float nm = 3.4e38f;
        #pragma unroll
        for (int t = 0; t < SAE_K; ++t) nm = fminf(nm, tv[t]);
        rmin = nm;
    }
}

// LDS layout (dynamic shared):
//   xs   : MBLK x D_DW   dwords (packed bf16 x-tile)      96 KB
//   lat  : MBLK x NCHUNK floats (latent chunk staging)    32 KB
//   topv : MBLK x SAE_K  floats                            5 KB
//   topi : MBLK x SAE_K  ints                              5 KB
#define XS_BYTES   (MBLK * D_DW * 4)
#define LAT_BYTES  (MBLK * NCHUNK * 4)
#define TV_BYTES   (MBLK * SAE_K * 4)
#define SMEM_BYTES (XS_BYTES + LAT_BYTES + 2 * TV_BYTES)

__global__ __launch_bounds__(256)
void sae_enc_topk(const float* __restrict__ x,
                  const float* __restrict__ W_enc,
                  const float* __restrict__ b_enc,
                  float* __restrict__ topv_out,
                  int*   __restrict__ topi_out)
{
    extern __shared__ char smem[];
    uint32_t* xs   = (uint32_t*)smem;
    float*    lat  = (float*)(smem + XS_BYTES);
    float*    topv = (float*)(smem + XS_BYTES + LAT_BYTES);
    int*      topi = (int*)  (smem + XS_BYTES + LAT_BYTES + TV_BYTES);

    const int tid   = threadIdx.x;
    const int lane  = tid & 31;
    const int wave  = tid >> 5;          // 0..7
    const int nsel  = lane & 15;         // WMMA column / M-row within half
    const int khalf = lane >> 4;         // 0: low K-half lanes, 1: high
    const int n0    = blockIdx.x * MBLK;

#if __has_builtin(__builtin_amdgcn_s_wait_tensorcnt)
    __builtin_amdgcn_s_wait_tensorcnt(0);   // harmless: no TDM ops outstanding
#endif

    // ---- stage x tile into LDS as packed bf16 (row-major, K-pairs per dword)
    for (int i = tid; i < MBLK * D_DW; i += 256) {
        const int row = i / D_DW, col = i - row * D_DW;
        const float2 f = ((const float2*)(x + (size_t)(n0 + row) * SAE_D))[col];
        xs[i] = pk_bf16(f.x, f.y);
    }
    if (tid < MBLK) {
        #pragma unroll
        for (int t = 0; t < SAE_K; ++t) {
            topv[tid * SAE_K + t] = -3.4e38f;
            topi[tid * SAE_K + t] = 0;
        }
    }
    __syncthreads();

    float rmin = -3.4e38f;   // current 20th-best for row 'tid' (tid < MBLK only)

    for (int chunk = 0; chunk < NCHUNKS; ++chunk) {
        const int l0 = chunk * NCHUNK + wave * 16;   // this wave's 16 latents

        // B-fragment source: one W_enc row per column lane, K-half by lane>=16
        const float* wrow = W_enc + (size_t)(l0 + nsel) * SAE_D + khalf * 16;

        // hoisted: bias load latency hides under the GEMM below
        const float bias = b_enc[l0 + nsel];

        // prefetch next chunk's weight rows into cache hierarchy
        if (chunk + 1 < NCHUNKS)
            __builtin_prefetch(wrow + (size_t)NCHUNK * SAE_D, 0, 1);

        v8f acc[4] = {};   // 4 M-tiles of 16x16 f32

        for (int kb = 0; kb < SAE_D; kb += 32) {
            // ---- B fragment: 16 consecutive f32 -> 8 packed bf16 dwords.
            // Compiler clauses the 4 global_load_b128 and pipelines them
            // across iterations via staggered s_wait_loadcnt.
            Frag16 bf;
            const float4* wp = (const float4*)(wrow + kb);
            #pragma unroll
            for (int q = 0; q < 4; ++q) {
                const float4 f = wp[q];
                bf.u[2 * q + 0] = pk_bf16(f.x, f.y);
                bf.u[2 * q + 1] = pk_bf16(f.z, f.w);
            }
            // ---- A fragments from LDS (2x ds_load_b128 each) + 4 WMMAs,
            // B reused across the 4 M-tiles
            Frag16 af[4];
            #pragma unroll
            for (int mt = 0; mt < 4; ++mt) {
                const uint4* ap =
                    (const uint4*)(xs + (mt * 16 + nsel) * D_DW + (kb >> 1));
                const uint4 q0 = ap[2 * khalf];       // K pairs 0..3  / 8..11
                const uint4 q1 = ap[2 * khalf + 1];   // K pairs 4..7 / 12..15
                af[mt].u[0] = q0.x; af[mt].u[1] = q0.y;
                af[mt].u[2] = q0.z; af[mt].u[3] = q0.w;
                af[mt].u[4] = q1.x; af[mt].u[5] = q1.y;
                af[mt].u[6] = q1.z; af[mt].u[7] = q1.w;
            }
            #pragma unroll
            for (int mt = 0; mt < 4; ++mt) {
                acc[mt] = __builtin_amdgcn_wmma_f32_16x16x32_bf16(
                    false, af[mt].v, false, bf.v, (short)0, acc[mt],
                    false, false);
            }
        }

        // ---- scatter C tiles (+bias) into latent staging chunk
        // C layout: VGPR r -> M=r (lanes 0-15), M=r+8 (lanes 16-31); N=lane&15
        #pragma unroll
        for (int mt = 0; mt < 4; ++mt) {
            #pragma unroll
            for (int r = 0; r < 8; ++r) {
                lat[(mt * 16 + khalf * 8 + r) * NCHUNK + wave * 16 + nsel] =
                    acc[mt][r] + bias;
            }
        }
        __syncthreads();

        // ---- top-K maintenance: one thread per row, float4 scan with
        //      vector-max early-out; insert path is rare (~130 hits / 12288)
        if (tid < MBLK) {
            const float4* lr4 = (const float4*)(lat + tid * NCHUNK);
            const int base = chunk * NCHUNK;
            float* tv = topv + tid * SAE_K;
            int*   ti = topi + tid * SAE_K;
            for (int j = 0; j < NCHUNK / 4; ++j) {
                const float4 v = lr4[j];
                const float hi = fmaxf(fmaxf(v.x, v.y), fmaxf(v.z, v.w));
                if (hi > rmin) {
                    topk_insert(v.x, base + 4 * j + 0, tv, ti, rmin);
                    topk_insert(v.y, base + 4 * j + 1, tv, ti, rmin);
                    topk_insert(v.z, base + 4 * j + 2, tv, ti, rmin);
                    topk_insert(v.w, base + 4 * j + 3, tv, ti, rmin);
                }
            }
        }
        __syncthreads();
    }

    // ---- emit top-K (values + indices) to workspace
    if (tid < MBLK) {
        #pragma unroll
        for (int t = 0; t < SAE_K; ++t) {
            topv_out[(size_t)(n0 + tid) * SAE_K + t] = topv[tid * SAE_K + t];
            topi_out[(size_t)(n0 + tid) * SAE_K + t] = topi[tid * SAE_K + t];
        }
    }
}

// ---------------------------------------------------------------------------
// Sparse decoder: one block per row n; 20 gathers per output element.
// W_dec[d, l] at d*L + l ; a d-row (48 KB) is contiguous, W_dec is L2-resident.
// FLOPs are negligible (2*N*K*D ~ 0.5 GFLOP) -> pure L2 gather kernel.
// ---------------------------------------------------------------------------
__global__ __launch_bounds__(256)
void sae_dec(const float* __restrict__ topv,
             const int*   __restrict__ topi,
             const float* __restrict__ W_dec,
             const float* __restrict__ b_dec,
             float* __restrict__ out)
{
    __shared__ float sv[SAE_K];
    __shared__ int   si[SAE_K];
    const int n = blockIdx.x;
    if (threadIdx.x < SAE_K) {
        sv[threadIdx.x] = topv[(size_t)n * SAE_K + threadIdx.x];
        si[threadIdx.x] = topi[(size_t)n * SAE_K + threadIdx.x];
    }
    __syncthreads();

    #pragma unroll 3
    for (int d = threadIdx.x; d < SAE_D; d += 256) {
        const float* wr = W_dec + (size_t)d * SAE_L;
        float acc = b_dec[d];
        #pragma unroll
        for (int j = 0; j < SAE_K; ++j)
            acc += sv[j] * wr[si[j]];
        out[(size_t)n * SAE_D + d] = acc;
    }
}

// ---------------------------------------------------------------------------
extern "C" void kernel_launch(void* const* d_in, const int* in_sizes, int n_in,
                              void* d_out, int out_size, void* d_ws, size_t ws_size,
                              hipStream_t stream)
{
    (void)in_sizes; (void)n_in; (void)out_size; (void)ws_size;

    const float* x     = (const float*)d_in[0];
    const float* W_enc = (const float*)d_in[1];
    const float* b_enc = (const float*)d_in[2];
    const float* W_dec = (const float*)d_in[3];
    const float* b_dec = (const float*)d_in[4];
    float* out = (float*)d_out;

    float* topv = (float*)d_ws;
    int*   topi = (int*)((char*)d_ws + (size_t)SAE_N * SAE_K * sizeof(float));

    sae_enc_topk<<<SAE_N / MBLK, 256, SMEM_BYTES, stream>>>(
        x, W_enc, b_enc, topv, topi);
    sae_dec<<<SAE_N, 256, 0, stream>>>(topv, topi, W_dec, b_dec, out);
}
